// CAMLocalHead_14018773254293
// MI455X (gfx1250) — matrix operand order repaired and checked by reading, not compiled, not run on hardware
//
#include <hip/hip_runtime.h>
#include <hip/hip_bf16.h>
#include <math.h>

// ---------- types ----------
typedef __bf16 v16bf __attribute__((ext_vector_type(16)));
typedef float  v8f   __attribute__((ext_vector_type(8)));
typedef unsigned u32x4 __attribute__((ext_vector_type(4)));
typedef int      i32x8 __attribute__((ext_vector_type(8)));
typedef int      i32x4 __attribute__((ext_vector_type(4)));

// fp32 -> bf16 (round to nearest even), raw u16
__device__ __forceinline__ unsigned f2bf(float f) {
    union { float f; unsigned u; } z; z.f = f;
    unsigned u = z.u;
    return ((u + 0x7FFFu + ((u >> 16) & 1u)) >> 16) & 0xFFFFu;
}

// ============================================================
// Kernel 0: zero the scalar accumulator
// ============================================================
__global__ void k_zero(float* out) {
    if (threadIdx.x == 0 && blockIdx.x == 0) out[0] = 0.0f;
}

// ============================================================
// Kernel 1: CAM target. One block per batch sample (B=8).
// ============================================================
__global__ __launch_bounds__(256)
void k_cam(const float* __restrict__ x,          // [8,2048,784]
           const float* __restrict__ pred,       // [8,400]
           const float* __restrict__ pw,         // [400,2048]
           float* __restrict__ cam_gt)           // [8,784] (ws)
{
    __shared__ float s_cam[784];
    __shared__ float s_r0[256];
    __shared__ float s_r1[256];
    __shared__ int   s_i0[256];
    __shared__ int   s_cls;

    const int b = blockIdx.x;
    const int tid = threadIdx.x;

    // argmax over 400 (sigmoid monotone)
    float bv = -3.4e38f; int bi = 0;
    for (int k = tid; k < 400; k += 256) {
        float v = pred[b * 400 + k];
        if (v > bv) { bv = v; bi = k; }
    }
    s_r0[tid] = bv; s_i0[tid] = bi;
    __syncthreads();
    for (int s = 128; s > 0; s >>= 1) {
        if (tid < s) {
            float ov = s_r0[tid + s]; int oi = s_i0[tid + s];
            if (ov > s_r0[tid] || (ov == s_r0[tid] && oi < s_i0[tid])) {
                s_r0[tid] = ov; s_i0[tid] = oi;
            }
        }
        __syncthreads();
    }
    if (tid == 0) s_cls = s_i0[0];
    __syncthreads();

    // cam[n] = dot(pw[cls,:], x[b,:,n])
    const float* wrow = pw + (size_t)s_cls * 2048;
    const float* xb   = x + (size_t)b * 2048 * 784;
    for (int n = tid; n < 784; n += 256) {
        float acc = 0.0f;
        for (int c = 0; c < 2048; ++c)
            acc = fmaf(wrow[c], xb[(size_t)c * 784 + n], acc);
        s_cam[n] = acc;
    }
    __syncthreads();

    // min/max
    float mn = 3.4e38f, mx = -3.4e38f;
    for (int n = tid; n < 784; n += 256) {
        float v = s_cam[n];
        mn = fminf(mn, v); mx = fmaxf(mx, v);
    }
    s_r0[tid] = mn; s_r1[tid] = mx;
    __syncthreads();
    for (int s = 128; s > 0; s >>= 1) {
        if (tid < s) {
            s_r0[tid] = fminf(s_r0[tid], s_r0[tid + s]);
            s_r1[tid] = fmaxf(s_r1[tid], s_r1[tid + s]);
        }
        __syncthreads();
    }
    const float cmin = s_r0[0], cmax = s_r1[0];
    const float inv = 1.0f / (cmax - cmin);
    __syncthreads();

    for (int n = tid; n < 784; n += 256)
        s_cam[n] = (s_cam[n] - cmin) * inv;
    __syncthreads();

    // stable rank: keep top 392 (matches lax.top_k tie-break)
    for (int n = tid; n < 784; n += 256) {
        const float v = s_cam[n];
        int rank = 0;
        for (int m = 0; m < 784; ++m) {
            float u = s_cam[m];
            rank += (u > v) || (u == v && m < n);
        }
        cam_gt[b * 784 + n] = (rank < 392) ? v : 0.0f;
    }
}

// ============================================================
// Kernel 2: repack conv1_w fp32 [512,2048,1,3,3] -> bf16,
// FRAGMENT-MAJOR: wq[j][kb][d][r], r=0..15 c-pairs of one 32-K
// step.  One lane's B fragment = 8 contiguous u32, and each
// (j,kb) block is one contiguous 32KB TDM transfer.
// ============================================================
__global__ __launch_bounds__(256)
void k_packw(const float* __restrict__ w, unsigned* __restrict__ wq, int total)
{
    int i = blockIdx.x * 256 + threadIdx.x;
    if (i >= total) return;
    int r  = i & 15;            // c-pair within K-step
    int d  = (i >> 4) & 511;    // output channel
    int kb = (i >> 13) & 63;    // K-step (32 channels each)
    int j  = i >> 19;           // kernel tap 0..8
    size_t off = (size_t)d * (2048 * 9) + (size_t)(kb * 32 + 2 * r) * 9 + j;
    unsigned lo = f2bf(w[off]);
    unsigned hi = f2bf(w[off + 9]);
    wq[i] = lo | (hi << 16);
}

// ============================================================
// TDM: async-load one 32KB fragment-major weight block into LDS,
// padded 4 DWORDs every 16 DWORDs => LDS rows of 20 u32 so every
// 8-DWORD fragment is 16B-aligned (ds_load_b128) and bank-free.
// 6-arg builtin form: (g0, g1, g2, g3, g4, cpol).
// D# group0: count=1 | lds_addr | global_addr(57b) | type=2
// D# group1: data_size=4B, pad_enable, pad_interval=16DW,
//            pad_amount=4DW, 1-D tile of 8192 DWORDs
// ============================================================
__device__ __forceinline__ void tdm_load_b(const unsigned* gsrc, unsigned lds_off)
{
    size_t ga = (size_t)gsrc;
    u32x4 g0;
    g0[0] = 1u;                                         // count = 1 (valid)
    g0[1] = lds_off;                                    // lds_addr (bytes)
    g0[2] = (unsigned)(ga & 0xFFFFFFFFu);               // global_addr[31:0]
    g0[3] = (unsigned)((ga >> 32) & 0x01FFFFFFu)        // global_addr[56:32]
          | 0x80000000u;                                // type = 2 (image)
    i32x8 g1;
    g1[0] = (int)((2u << 16)        // data_size = 4 bytes
                | (1u << 20)        // pad_enable
                | (3u << 22)        // pad_interval: 16 DWORDs
                | (3u << 25));      // pad_amount:   4 DWORDs (code 3)
    g1[1] = (int)(0x2000u << 16);   // tensor_dim0 = 8192 (bits 79:48, low half)
    g1[2] = (int)(1u << 16);        // tensor_dim0 hi = 0 ; tensor_dim1 = 1
    g1[3] = (int)(0x2000u << 16);   // tensor_dim1 hi = 0 ; tile_dim0 = 8192
    g1[4] = 1;                      // tile_dim1 = 1, tile_dim2 = 0
    g1[5] = 8192;                   // tensor_dim0_stride (lo)
    g1[6] = 0;                      // stride hi / tensor_dim1_stride lo
    g1[7] = 0;
    i32x4 z4 = {0, 0, 0, 0};                     // groups 2/3 unused (<=2D tensor)
    i32x8 z8 = {0, 0, 0, 0, 0, 0, 0, 0};         // trailing group (zero-filled)
    __builtin_amdgcn_tensor_load_to_lds(g0, g1, z4, z4, z8, 0);
}

// ============================================================
// Kernel 3: fused implicit-GEMM conv (bf16 WMMA, f32 acc)
//           + bias + ReLU + score dot + BCEWithLogits mean.
// 512 threads (16 waves); tile 128 pixels x 512 chans, full K.
// Weight tiles double-buffered in LDS via TDM (overlaps compute).
// ============================================================
#define BROW 20            // padded row stride (u32): 16 data + 4 pad
#define BBUF (512 * BROW)  // 10240 u32 = 40KB per buffer
#define NITER (9 * 64)     // taps * K-steps

__global__ __launch_bounds__(512)
void k_conv_loss(const float*    __restrict__ x,       // [8,2048,16,7,7]
                 const unsigned* __restrict__ wq,      // [9,64,512,16] u32
                 const float*    __restrict__ bias,    // [512]
                 const float*    __restrict__ sw,      // [512]
                 const float*    __restrict__ sb,      // [1]
                 const float*    __restrict__ cam_gt,  // [6272]
                 float*          __restrict__ out)     // [1]
{
    __shared__ __align__(16) unsigned sA[128][16];   // im2col tile (8KB)
    __shared__ __align__(16) unsigned sBq[2][BBUF];  // weight tiles (80KB)
    __shared__ float sLogit[128];

    const int tid  = threadIdx.x;
    const int wv   = tid >> 5;           // wave 0..15
    const int lane = tid & 31;
    const int g    = lane >> 4;
    const int ln   = lane & 15;
    const int wr   = wv & 3;             // m-group: mtiles {2wr, 2wr+1}
    const int wc   = wv >> 2;            // n-group: ntiles {8wc..8wc+7}
    const int pBase = blockIdx.x * 128;

    v8f acc[2][8];
    #pragma unroll
    for (int mi = 0; mi < 2; ++mi)
        #pragma unroll
        for (int nc = 0; nc < 8; ++nc)
            acc[mi][nc] = (v8f)0.0f;

    // prologue: prefetch weight block 0 into buffer 0
    if (wv == 0)
        tdm_load_b(wq, (unsigned)(size_t)&sBq[0][0]);

    for (int it = 0; it < NITER; ++it) {
        const int j  = it >> 6;          // tap 0..8
        const int c0 = (it & 63) * 32;   // channel base
        const int kh = j / 3, kw = j % 3;

        // ---- stage A tile: im2col + fp32->bf16 pair pack (8KB)
        #pragma unroll
        for (int s = 0; s < 4; ++s) {
            int idx = tid + s * 512;
            int p  = idx >> 4;
            int cp = idx & 15;
            int np = pBase + p;
            int w_ = np % 7;
            int h_ = (np / 7) % 7;
            int t_ = (np / 49) & 15;
            int b_ = np / 784;
            int hh = h_ + kh - 1, ww = w_ + kw - 1;
            unsigned val = 0;
            if (hh >= 0 && hh < 7 && ww >= 0 && ww < 7) {
                size_t base = ((((size_t)b_ * 2048 + (c0 + 2 * cp)) * 16 + t_) * 7 + hh) * 7 + ww;
                unsigned lo = f2bf(x[base]);
                unsigned hi = f2bf(x[base + 784]);   // c+1 (stride T*H*W)
                val = lo | (hi << 16);
            }
            sA[p][cp] = val;
        }

        // ---- wave 0: prefetch next weight block, wait for current one
        if (wv == 0) {
            int nx = (it + 1 == NITER) ? 0 : it + 1;   // wrap: harmless tail prefetch
            tdm_load_b(wq + (size_t)nx * 8192,
                       (unsigned)(size_t)&sBq[(it + 1) & 1][0]);
            __builtin_amdgcn_s_wait_tensorcnt((short)1);  // block i resident
        }
        __syncthreads();   // publish sA + sBq[it&1] to all waves

        const unsigned* bbuf = &sBq[it & 1][0];

        // ---- A fragments (2 m-tiles): 2x ds_load_b128 each
        v16bf aF[2];
        #pragma unroll
        for (int mi = 0; mi < 2; ++mi) {
            int m = (wr * 2 + mi) * 16 + ln;
            union { unsigned u[8]; v16bf v; } ua;
            #pragma unroll
            for (int v = 0; v < 8; ++v) {
                int cp = v + (g << 3);
                ua.u[v] = sA[m][cp];
            }
            aF[mi] = ua.v;
        }
        // ---- 8 n-tiles x 2 m-tiles of WMMA; B frag = 2x ds_load_b128
        #pragma unroll
        for (int nc = 0; nc < 8; ++nc) {
            int dcol = (wc * 8 + nc) * 16 + ln;
            const unsigned* bp = bbuf + dcol * BROW + (g << 3);   // 16B aligned
            union { unsigned u[8]; v16bf v; } ub;
            #pragma unroll
            for (int v = 0; v < 8; ++v) ub.u[v] = bp[v];
            #pragma unroll
            for (int mi = 0; mi < 2; ++mi) {
                acc[mi][nc] = __builtin_amdgcn_wmma_f32_16x16x32_bf16(
                    false, aF[mi], false, ub.v, (short)0, acc[mi][nc],
                    false, false);
            }
        }
        __syncthreads();   // before overwriting sA next iter
    }

    // ---- epilogue: bias + ReLU + score dot, reduce over d per pixel
    if (tid < 128) sLogit[tid] = 0.0f;
    __syncthreads();
    #pragma unroll
    for (int mi = 0; mi < 2; ++mi) {
        #pragma unroll
        for (int r = 0; r < 8; ++r) {
            int p = (wr * 2 + mi) * 16 + r + g * 8;   // C/D: M = r + g*8, N = ln
            float s = 0.0f;
            #pragma unroll
            for (int nc = 0; nc < 8; ++nc) {
                int d = (wc * 8 + nc) * 16 + ln;
                float v = acc[mi][nc][r] + bias[d];
                v = fmaxf(v, 0.0f);
                s = fmaf(v, sw[d], s);
            }
            atomicAdd(&sLogit[p], s);   // ds_add_f32
        }
    }
    __syncthreads();

    // ---- BCE with logits (pos_weight=1), mean-scaled
    float part = 0.0f;
    if (tid < 128) {
        float xl = sLogit[tid] + sb[0];
        float y  = cam_gt[pBase + tid];
        float t  = -xl;
        float spv;
        if (t > 20.0f)       spv = t;
        else if (t < -20.0f) spv = expf(t);
        else                 spv = log1pf(expf(t));
        part = ((1.0f - y) * xl + spv) * (1.0f / 6272.0f);
    }
    float* red = (float*)&sA[0][0];   // reuse sA as reduce scratch
    red[tid] = part;
    __syncthreads();
    for (int s = 256; s > 0; s >>= 1) {
        if (tid < s) red[tid] += red[tid + s];
        __syncthreads();
    }
    if (tid == 0) atomicAdd(out, red[0]);
}

// ============================================================
// Host launcher
// ============================================================
extern "C" void kernel_launch(void* const* d_in, const int* in_sizes, int n_in,
                              void* d_out, int out_size, void* d_ws, size_t ws_size,
                              hipStream_t stream) {
    const float* x      = (const float*)d_in[0];   // [8,2048,16,7,7]
    const float* pred   = (const float*)d_in[1];   // [8,400]
    const float* pw     = (const float*)d_in[2];   // [400,2048]
    const float* conv1w = (const float*)d_in[3];   // [512,2048,1,3,3]
    const float* conv1b = (const float*)d_in[4];   // [512]
    const float* scorew = (const float*)d_in[5];   // [512]
    const float* scoreb = (const float*)d_in[6];   // [1]
    float* out = (float*)d_out;

    float*    ws_cam = (float*)d_ws;                         // 6272 f32
    unsigned* ws_wq  = (unsigned*)((char*)d_ws + 6272 * 4);  // 9*64*512*16 u32
    const int wq_total = 9 * 64 * 512 * 16;

    k_zero<<<1, 64, 0, stream>>>(out);
    k_cam<<<8, 256, 0, stream>>>(x, pred, pw, ws_cam);
    k_packw<<<(wq_total + 255) / 256, 256, 0, stream>>>(conv1w, ws_wq, wq_total);
    k_conv_loss<<<49, 512, 0, stream>>>(x, ws_wq, conv1b, scorew, scoreb,
                                        ws_cam, out);
}
